// CrossAttentionWithMask_73160472920340
// MI455X (gfx1250) — matrix operand order repaired and checked by reading, not compiled, not run on hardware
//
#include <hip/hip_runtime.h>
#include <hip/hip_bf16.h>
#include <math.h>

// ---------------------------------------------------------------------------
// CrossAttentionWithMask for MI455X (gfx1250, wave32, WMMA, TDM).
//
// Shapes: query [1,256,64,64] f32, support [4,256,64,64] f32,
//         W* [256,256], b* [256].  N=4, C=256, HW=4096.
//   proj_kernel : bf16 WMMA GEMM  X @ W^T + b  -> bf16 Q/K/V in workspace
//                 (V stored channel-major so attention V B-frags are
//                  contiguous in LDS).
//   attn_kernel : fused flash-attention.  Per workgroup: 4 waves, 64 queries,
//                 loop over 4096 keys in 32-key LDS tiles staged by the
//                 Tensor Data Mover (tensor_load_to_lds + s_wait_tensorcnt).
//                 Online softmax; max(probs) == 1/row_sum -> free gate.
// Matrix math: v_wmma_f32_16x16x32_bf16 (fp32 accumulate).
// ---------------------------------------------------------------------------

typedef __attribute__((ext_vector_type(16))) __bf16 v16bf;
typedef __attribute__((ext_vector_type(8)))  float  v8f;
typedef __attribute__((ext_vector_type(4)))  unsigned int v4u;
typedef __attribute__((ext_vector_type(8)))  int   v8i;
typedef __attribute__((ext_vector_type(4)))  int   v4i;

#define DIMC 256
#define HW   4096
#define NSUP 4

// A-frag k order chosen here (contiguous; B frags use the same).  lane half h=lane/16, vgpr v holds
// K = h*16 + v*2  (and +1).  M = lane%16 for all lanes.
__device__ __forceinline__ int a_kofs(int h, int v) {
    return h * 16 + v * 2;
}

// ---------------------------------------------------------------------------
// Tensor-DMA descriptor builders (CDNA5 ISA ch.8, D# groups 0/1).
// 2D tile of bf16 (data_size code 1): tile_d0 elems/row, tile_d1 rows,
// row stride = stride0 elems.  LDS destination packed contiguously.
// ---------------------------------------------------------------------------
__device__ __forceinline__ v4u tdm_g0(unsigned lds_off, unsigned long long ga) {
    v4u g;
    g[0] = 1u;                                           // count=1, user mode
    g[1] = lds_off;                                      // lds_addr [63:32]
    g[2] = (unsigned)(ga & 0xFFFFFFFFu);                 // global_addr[31:0]
    g[3] = (unsigned)((ga >> 32) & 0x1FFFFFFu)           // global_addr[56:32]
         | (2u << 30);                                   // type = 2 (image)
    return g;
}
__device__ __forceinline__ v8i tdm_g1(unsigned tensor_d0, unsigned tensor_d1,
                                      unsigned tile_d0, unsigned tile_d1,
                                      unsigned long long stride0) {
    v8i g;
    g[0] = (int)(1u << 16);                              // data_size=1 (2B)
    g[1] = (int)((tensor_d0 & 0xFFFFu) << 16);           // tensor_dim0[15:0]
    g[2] = (int)((tensor_d0 >> 16) | ((tensor_d1 & 0xFFFFu) << 16));
    g[3] = (int)((tensor_d1 >> 16) | (tile_d0 << 16));   // tile_dim0
    g[4] = (int)(tile_d1 & 0xFFFFu);                     // tile_dim1 (dim2=0)
    g[5] = (int)(stride0 & 0xFFFFFFFFu);                 // dim0_stride[31:0]
    g[6] = (int)((stride0 >> 32) & 0xFFFFu);             // dim0_stride[47:32]
    g[7] = 0;
    return g;
}

__device__ __forceinline__ void tdm_load_tile(unsigned lds_off,
                                              unsigned long long gaddr,
                                              unsigned tensor_d0, unsigned tensor_d1,
                                              unsigned tile_d0, unsigned tile_d1,
                                              unsigned long long stride0) {
#if __has_builtin(__builtin_amdgcn_tensor_load_to_lds)
    const v4u g0 = tdm_g0(lds_off, gaddr);
    const v8i g1 = tdm_g1(tensor_d0, tensor_d1, tile_d0, tile_d1, stride0);
    const v4i z4 = {};
#if __clang_major__ >= 23
    const v8i z8 = {};
    __builtin_amdgcn_tensor_load_to_lds(g0, g1, z4, z4, z8, 0);
#else
    __builtin_amdgcn_tensor_load_to_lds(g0, g1, z4, z4, 0);
#endif
#endif
}

// ===========================================================================
// Projection: out[row,col] = sum_c X[row,c] * W[col,c] + b[col]
// X element (row=hw, c) lives at in[c*HW + hw]  (input is [C,H,W]).
// mode 0: Q (row-major out), 1: K (row-major out), 2: V (out[col*HW+row]).
// One wave per 16x16 output tile; 8 bf16 WMMAs over C=256.
// ===========================================================================
__global__ __launch_bounds__(128) void proj_kernel(
    const float* __restrict__ X, const float* __restrict__ W,
    const float* __restrict__ bias, __bf16* __restrict__ out, int mode)
{
    const int n    = blockIdx.y;
    const float* Xn = X + (size_t)n * DIMC * HW;   // mode 0 -> gridDim.y==1
    const int wave = threadIdx.x >> 5;
    const int lane = threadIdx.x & 31;
    const int h    = lane >> 4;         // lane half
    const int ln   = lane & 15;         // lane-in-half

    const int tile = blockIdx.x * 4 + wave;        // 4096 tiles total
    const int rowb = (tile >> 4) * 16;             // 256 row tiles
    const int colb = (tile & 15) * 16;             // 16  col tiles

    // D init = bias broadcast along rows: D[M, N=ln] = b[colb+ln]
    const float b0 = bias[colb + ln];
    v8f acc;
#pragma unroll
    for (int r = 0; r < 8; ++r) acc[r] = b0;

    const int hw = rowb + ln;                      // A-matrix row (M = ln)
#pragma unroll
    for (int cc = 0; cc < 8; ++cc) {
        const int cb = cc * 32;
        v16bf a, bm;
        // A frag: X[hw, cb+k] = Xn[(cb+k)*HW + hw]
#pragma unroll
        for (int v = 0; v < 8; ++v) {
            const int k = a_kofs(h, v);
            a[2 * v]     = (__bf16)Xn[(size_t)(cb + k)     * HW + hw];
            a[2 * v + 1] = (__bf16)Xn[(size_t)(cb + k + 1) * HW + hw];
        }
        // B frag (32x16): B[k=c][n=col] = W[colb+ln, c]; lane holds
        // K = h*16 + 2v (+1): 16 contiguous c values from W's row.
        const float* wr = W + (size_t)(colb + ln) * DIMC + cb + h * 16;
#pragma unroll
        for (int v = 0; v < 8; ++v) {
            bm[2 * v]     = (__bf16)wr[2 * v];
            bm[2 * v + 1] = (__bf16)wr[2 * v + 1];
        }
        acc = __builtin_amdgcn_wmma_f32_16x16x32_bf16(
                  false, a, false, bm, (short)0, acc, false, false);
    }

    // C/D layout: vgpr r, lane -> (M = r + 8*h, N = ln)
    __bf16* on = out + (size_t)n * HW * DIMC;
#pragma unroll
    for (int r = 0; r < 8; ++r) {
        const int row = rowb + r + 8 * h;
        const int col = colb + ln;
        const __bf16 v = (__bf16)acc[r];
        if (mode == 2) on[(size_t)col * HW + row] = v;    // V: channel-major
        else           on[(size_t)row * DIMC + col] = v;  // Q/K: row-major
    }
}

// ===========================================================================
// Fused attention.  Block = 128 threads (4 waves); wave w owns 16 queries.
// K/V tiles staged into LDS by the Tensor Data Mover each iteration.
// ===========================================================================
__global__ __launch_bounds__(128) void attn_kernel(
    const __bf16* __restrict__ Qb,   // [HW, C]
    const __bf16* __restrict__ Kb,   // [N, HW, C]
    const __bf16* __restrict__ Vt,   // [N, C, HW]  (channel-major)
    const float* __restrict__ thr_p, const float* __restrict__ temp_p,
    float* __restrict__ out)         // [N, C, HW]
{
    __shared__ __align__(32) __bf16 Qs[64 * DIMC];     // 32 KB
    __shared__ __align__(32) __bf16 Ks[32 * DIMC];     // 16 KB (key-major)
    __shared__ __align__(32) __bf16 Vs[DIMC * 32];     // 16 KB (chan-major)
    __shared__ __align__(32) __bf16 Ptmp[4][16 * 32];  //  4 KB P transpose

    const int n    = blockIdx.x >> 6;
    const int qblk = blockIdx.x & 63;                  // 64 queries / block
    const int tid  = threadIdx.x;
    const int wave = tid >> 5;
    const int lane = tid & 31;
    const int h    = lane >> 4;
    const int ln   = lane & 15;

    // ---- stage this block's Q rows (64 x 256 bf16) into LDS -------------
    {
        const uint4* src = (const uint4*)(Qb + (size_t)qblk * 64 * DIMC);
        uint4* dst = (uint4*)Qs;
#pragma unroll
        for (int i = 0; i < 16; ++i) dst[tid + i * 128] = src[tid + i * 128];
    }

    const __bf16* Kg = Kb + (size_t)n * HW * DIMC;
    const __bf16* Vg = Vt + (size_t)n * DIMC * HW;
    const unsigned ldsK = (unsigned)(size_t)&Ks[0];    // LDS byte offsets
    const unsigned ldsV = (unsigned)(size_t)&Vs[0];
    const float scale = 0.0625f;                       // 1/sqrt(256)

    v8f O[16];                                         // 16 queries x 256 ch
#pragma unroll
    for (int t = 0; t < 16; ++t) O[t] = v8f{};
    float m[8], lsum[8];
#pragma unroll
    for (int r = 0; r < 8; ++r) { m[r] = -INFINITY; lsum[r] = 0.0f; }

    for (int kt = 0; kt < HW / 32; ++kt) {
        __syncthreads();                               // LDS tile reuse
        if (wave == 0) {
            // K tile: 32 contiguous key rows -> 1 line of 8192 bf16.
            tdm_load_tile(ldsK,
                          (unsigned long long)(size_t)(Kg + (size_t)kt * 32 * DIMC),
                          HW * DIMC, 1, 32 * DIMC, 1, HW * DIMC);
            // V tile: 256 channel rows x 32 keys, row stride HW elems.
            tdm_load_tile(ldsV,
                          (unsigned long long)(size_t)(Vg + (size_t)kt * 32),
                          HW, DIMC, 32, DIMC, HW);
            __builtin_amdgcn_s_wait_tensorcnt(0);
        }
        if (kt + 1 < HW / 32)                          // global_prefetch_b8
            __builtin_prefetch(Kg + (size_t)(kt + 1) * 32 * DIMC, 0, 1);
        __syncthreads();                               // tiles visible to all

        // ---- S = Q K^T for this wave's 16 queries vs 32 keys -----------
        v8f S0 = v8f{}, S1 = v8f{};
        const __bf16* qr = &Qs[(wave * 16 + ln) * DIMC];
#pragma unroll
        for (int cc = 0; cc < 8; ++cc) {
            v16bf a;
#pragma unroll
            for (int v = 0; v < 8; ++v) {
                const int k = cc * 32 + a_kofs(h, v);
                a[2 * v] = qr[k]; a[2 * v + 1] = qr[k + 1];
            }
            // B frag: B[k=c][n=key]; lane holds 16 contiguous c of key row
            const v16bf b0 = *(const v16bf*)&Ks[ln * DIMC + cc * 32 + h * 16];
            S0 = __builtin_amdgcn_wmma_f32_16x16x32_bf16(
                     false, a, false, b0, (short)0, S0, false, false);
            const v16bf b1 = *(const v16bf*)&Ks[(16 + ln) * DIMC + cc * 32 + h * 16];
            S1 = __builtin_amdgcn_wmma_f32_16x16x32_bf16(
                     false, a, false, b1, (short)0, S1, false, false);
        }

        // ---- online softmax (row = r + 8*h; reduce over 16-lane half) --
        float alpha[8];
#pragma unroll
        for (int r = 0; r < 8; ++r) {
            const float s0 = S0[r] * scale, s1 = S1[r] * scale;
            float mx = fmaxf(s0, s1);
#pragma unroll
            for (int off = 1; off < 16; off <<= 1)
                mx = fmaxf(mx, __shfl_xor(mx, off, 32));
            const float mn = fmaxf(m[r], mx);
            alpha[r] = __expf(m[r] - mn);
            const float p0 = __expf(s0 - mn), p1 = __expf(s1 - mn);
            float rs = p0 + p1;
#pragma unroll
            for (int off = 1; off < 16; off <<= 1)
                rs += __shfl_xor(rs, off, 32);
            lsum[r] = lsum[r] * alpha[r] + rs;
            m[r] = mn;
            // transpose P through per-wave LDS scratch (same-wave ordering)
            const int M = r + 8 * h;
            Ptmp[wave][M * 32 + ln]      = (__bf16)p0;
            Ptmp[wave][M * 32 + 16 + ln] = (__bf16)p1;
        }
#pragma unroll
        for (int t = 0; t < 16; ++t)
#pragma unroll
            for (int r = 0; r < 8; ++r) O[t][r] *= alpha[r];

        // ---- P A-frag (16 queries x 32 keys) ---------------------------
        v16bf pa;
        const __bf16* pr = &Ptmp[wave][ln * 32];
#pragma unroll
        for (int v = 0; v < 8; ++v) {
            const int k = a_kofs(h, v);
            pa[2 * v] = pr[k]; pa[2 * v + 1] = pr[k + 1];
        }
        // ---- O += P V ; B[k=key][n=ch] from channel-major Vs -----------
#pragma unroll
        for (int t = 0; t < 16; ++t) {
            const v16bf bv = *(const v16bf*)&Vs[(t * 16 + ln) * 32 + h * 16];
            O[t] = __builtin_amdgcn_wmma_f32_16x16x32_bf16(
                       false, pa, false, bv, (short)0, O[t], false, false);
        }
    }

    // ---- finalize: attended/row_sum, gate = sigmoid((1/lsum - thr)*temp)
    const float thr  = 1.0f / (1.0f + __expf(-thr_p[0]));    // sigmoid
    const float temp = log1pf(__expf(temp_p[0]));            // softplus
    float* outg = out + (size_t)n * DIMC * HW + qblk * 64 + wave * 16;
#pragma unroll
    for (int r = 0; r < 8; ++r) {
        const float inv  = 1.0f / lsum[r];                   // == max(probs)
        const float gate = 1.0f / (1.0f + __expf(-(inv - thr) * temp));
        const float g = inv * gate;
        const int q = r + 8 * h;
#pragma unroll
        for (int t = 0; t < 16; ++t)
            outg[(size_t)(t * 16 + ln) * HW + q] = O[t][r] * g;
    }
}

// ===========================================================================
extern "C" void kernel_launch(void* const* d_in, const int* in_sizes, int n_in,
                              void* d_out, int out_size, void* d_ws, size_t ws_size,
                              hipStream_t stream)
{
    const float* query   = (const float*)d_in[0];
    const float* support = (const float*)d_in[1];
    const float* Wq = (const float*)d_in[2];
    const float* bq = (const float*)d_in[3];
    const float* Wk = (const float*)d_in[4];
    const float* bk = (const float*)d_in[5];
    const float* Wv = (const float*)d_in[6];
    const float* bv = (const float*)d_in[7];
    const float* thr  = (const float*)d_in[8];
    const float* temp = (const float*)d_in[9];

    // workspace: Qb 2MB | Kb 8MB | Vt 8MB  (bf16)
    __bf16* Qb = (__bf16*)d_ws;
    __bf16* Kb = Qb + (size_t)HW * DIMC;
    __bf16* Vt = Kb + (size_t)NSUP * HW * DIMC;

    dim3 blk(128);
    proj_kernel<<<dim3(1024, 1),    blk, 0, stream>>>(query,   Wq, bq, Qb, 0);
    proj_kernel<<<dim3(1024, NSUP), blk, 0, stream>>>(support, Wk, bk, Kb, 1);
    proj_kernel<<<dim3(1024, NSUP), blk, 0, stream>>>(support, Wv, bv, Vt, 2);
    attn_kernel<<<dim3(NSUP * 64),  blk, 0, stream>>>(Qb, Kb, Vt, thr, temp,
                                                      (float*)d_out);
}
